// MoELayer_5007931867524
// MI455X (gfx1250) — compile-verified, matrix operand order checked by reading, not run on hardware
//
#include <hip/hip_runtime.h>
#include <math.h>

// Problem dims (fixed by the reference)
#define TOKENS 8192   // B*S
#define DDIM   1024
#define HDIM   4096
#define NEXP   8

// --- CDNA5 WMMA types ---------------------------------------------------
typedef __attribute__((ext_vector_type(16))) __bf16 v16bf;
typedef __attribute__((ext_vector_type(8)))  float  v8f;
typedef __attribute__((ext_vector_type(4)))  unsigned int u32x4;
typedef __attribute__((ext_vector_type(4)))  float  f32x4;

union FragAB { v16bf v; u32x4 q[2]; };

__device__ __forceinline__ unsigned short f32_to_bf16(float f) {
  __bf16 b = (__bf16)f;                    // native v_cvt path on gfx1250
  return __builtin_bit_cast(unsigned short, b);
}

// --- fp32 -> bf16 conversion of x --------------------------------------
__global__ void __launch_bounds__(256) moe_convert(const float* __restrict__ src,
                                                   unsigned short* __restrict__ dst,
                                                   int n) {
  for (int i = blockIdx.x * blockDim.x + threadIdx.x; i < n;
       i += gridDim.x * blockDim.x)
    dst[i] = f32_to_bf16(src[i]);
}

// --- gating: logits -> softmax -> top-2 raw probs scattered dense -------
__global__ void __launch_bounds__(256) moe_gate(const float* __restrict__ x,
                                                const float* __restrict__ Wg,
                                                const float* __restrict__ bg,
                                                float* __restrict__ wdense) {
  const int lane  = threadIdx.x & 31;
  const int token = blockIdx.x * 8 + (threadIdx.x >> 5);

  float acc[NEXP];
#pragma unroll
  for (int e = 0; e < NEXP; ++e) acc[e] = 0.f;

  const float* xr = x + (size_t)token * DDIM;
  for (int d = lane; d < DDIM; d += 32) {
    const float xv = xr[d];
    const float* wr = Wg + (size_t)d * NEXP;
#pragma unroll
    for (int e = 0; e < NEXP; ++e) acc[e] += xv * wr[e];
  }
#pragma unroll
  for (int e = 0; e < NEXP; ++e) {
#pragma unroll
    for (int off = 16; off >= 1; off >>= 1)
      acc[e] += __shfl_xor(acc[e], off, 32);
  }

  if (lane == 0) {
    float lg[NEXP];
    float m = -1e30f;
#pragma unroll
    for (int e = 0; e < NEXP; ++e) { lg[e] = acc[e] + bg[e]; m = fmaxf(m, lg[e]); }
    float p[NEXP], s = 0.f;
#pragma unroll
    for (int e = 0; e < NEXP; ++e) { p[e] = __expf(lg[e] - m); s += p[e]; }
    const float inv = 1.f / s;
#pragma unroll
    for (int e = 0; e < NEXP; ++e) p[e] *= inv;
    // top-2, first-index wins ties (matches jax.lax.top_k)
    int i1 = 0;
#pragma unroll
    for (int e = 1; e < NEXP; ++e) if (p[e] > p[i1]) i1 = e;
    int i2 = (i1 == 0) ? 1 : 0;
#pragma unroll
    for (int e = 0; e < NEXP; ++e)
      if (e != i1 && e != i2 && p[e] > p[i2]) i2 = e;
    float* wr = wdense + (size_t)token * NEXP;
#pragma unroll
    for (int e = 0; e < NEXP; ++e)
      wr[e] = (e == i1 || e == i2) ? p[e] : 0.f;
  }
}

// --- shared WMMA mainloop: C[128x128] += A[128xK](bf16) * B[Kx128](f32) --
// A is bf16 row-major with lda == Kdim, staged to LDS via
// GLOBAL_LOAD_ASYNC_TO_LDS_B128 (ASYNCcnt). B is f32 row-major [Kdim x N],
// converted to bf16 and transposed into LDS as Blds[n][k].
#define BK   32
#define SK   40          // padded LDS row stride (ushorts)

__device__ __forceinline__ void gemm_mainloop(
    const unsigned short* __restrict__ A, int Kdim,
    const float* __restrict__ Bg, int N,
    int blockRow, int blockCol,
    unsigned short* Alds, unsigned short* Blds,
    v8f acc[2][4]) {
  const int tid   = threadIdx.x;
  const int lane  = tid & 31;
  const int wid   = tid >> 5;
  const int wm    = wid & 3;        // wave row group (32 rows each)
  const int wn    = wid >> 2;       // wave col group (64 cols each)
  const int lrow  = lane & 15;
  const int kbase = (lane >> 4) * 8;

  const int arow = tid >> 1;        // 0..127
  const int acb  = (tid & 1) * 16;  // 0 or 16
  const unsigned short* aAg = A + (size_t)(blockRow + arow) * Kdim + acb;
  // LDS destination byte address: generic shared pointers keep the LDS
  // byte offset in addr[31:0] (aperture layout), valid for async-to-LDS.
  const unsigned aLds = (unsigned)(size_t)&Alds[arow * SK + acb];

  for (int k0 = 0; k0 < Kdim; k0 += BK) {
    // ---- stage A tile: DMA 32 contiguous bytes/lane straight into LDS.
    // INST_OFFSET applies to both LDS and global addresses, so offset:16
    // places the second 8 bf16 right after the first in both spaces.
    {
      const unsigned short* ag = aAg + k0;
      asm volatile(
          "global_load_async_to_lds_b128 %0, %1, off\n\t"
          "global_load_async_to_lds_b128 %0, %1, off offset:16"
          :
          : "v"(aLds), "v"(ag)
          : "memory");
    }

    // ---- stage B tile: read f32 coalesced, convert, transpose into LDS
#pragma unroll
    for (int c = 0; c < 4; ++c) {
      const int lin = c * 1024 + tid * 4;
      const int kk = lin >> 7;      // 0..31
      const int nn = lin & 127;     // multiple of 4
      const f32x4 f = *(const f32x4*)(Bg + (size_t)(k0 + kk) * N + blockCol + nn);
      Blds[(nn + 0) * SK + kk] = f32_to_bf16(f.x);
      Blds[(nn + 1) * SK + kk] = f32_to_bf16(f.y);
      Blds[(nn + 2) * SK + kk] = f32_to_bf16(f.z);
      Blds[(nn + 3) * SK + kk] = f32_to_bf16(f.w);
    }
    if (k0 + BK < Kdim) {
      __builtin_prefetch(aAg + k0 + BK, 0, 1);                         // global_prefetch_b8
      __builtin_prefetch(Bg + (size_t)(k0 + BK) * N + blockCol + tid, 0, 1);
    }
    asm volatile("s_wait_asynccnt 0x0" ::: "memory");  // my async copies landed
    __syncthreads();                                   // everyone's landed

    // ---- fragments: 16-bit A layout => two 16B LDS loads per frag
    FragAB af[2], bf[4];
#pragma unroll
    for (int r = 0; r < 2; ++r) {
      const int base = (wm * 32 + r * 16 + lrow) * SK + kbase;
      af[r].q[0] = *(const u32x4*)&Alds[base];
      af[r].q[1] = *(const u32x4*)&Alds[base + 16];
    }
#pragma unroll
    for (int c = 0; c < 4; ++c) {
      const int base = (wn * 64 + c * 16 + lrow) * SK + kbase;
      bf[c].q[0] = *(const u32x4*)&Blds[base];
      bf[c].q[1] = *(const u32x4*)&Blds[base + 16];
    }
#pragma unroll
    for (int r = 0; r < 2; ++r)
#pragma unroll
      for (int c = 0; c < 4; ++c)
        acc[r][c] = __builtin_amdgcn_wmma_f32_16x16x32_bf16(
            false, af[r].v, false, bf[c].v, (short)0, acc[r][c], false, false);
    __syncthreads();
  }
}

// --- GEMM1: h = relu(x @ W1[e] + b1[e]) -> bf16 -------------------------
__global__ void __launch_bounds__(256) moe_gemm1(
    const unsigned short* __restrict__ xb,
    const float* __restrict__ W1e,
    const float* __restrict__ b1e,
    unsigned short* __restrict__ hb) {
  __shared__ unsigned short Alds[128 * SK];
  __shared__ unsigned short Blds[128 * SK];
  v8f acc[2][4] = {};
  const int blockRow = blockIdx.y * 128;
  const int blockCol = blockIdx.x * 128;
  gemm_mainloop(xb, DDIM, W1e, HDIM, blockRow, blockCol, Alds, Blds, acc);

  const int lane = threadIdx.x & 31, wid = threadIdx.x >> 5;
  const int wm = wid & 3, wn = wid >> 2;
  const int lrow = lane & 15, lhalf = lane >> 4;
#pragma unroll
  for (int r = 0; r < 2; ++r)
#pragma unroll
    for (int c = 0; c < 4; ++c) {
      const int col  = blockCol + wn * 64 + c * 16 + lrow;
      const float bias = b1e[col];
      const int row0 = blockRow + wm * 32 + r * 16 + lhalf * 8;
#pragma unroll
      for (int i = 0; i < 8; ++i) {
        float v = acc[r][c][i] + bias;
        v = v > 0.f ? v : 0.f;
        hb[(size_t)(row0 + i) * HDIM + col] = f32_to_bf16(v);
      }
    }
}

// --- GEMM2: out (+)= w_e * (h @ W2[e] + b2[e]) --------------------------
__global__ void __launch_bounds__(256) moe_gemm2(
    const unsigned short* __restrict__ hb,
    const float* __restrict__ W2e,
    const float* __restrict__ b2e,
    const float* __restrict__ wdense,
    int expert, int accum,
    float* __restrict__ out) {
  __shared__ unsigned short Alds[128 * SK];
  __shared__ unsigned short Blds[128 * SK];
  v8f acc[2][4] = {};
  const int blockRow = blockIdx.y * 128;
  const int blockCol = blockIdx.x * 128;
  gemm_mainloop(hb, HDIM, W2e, DDIM, blockRow, blockCol, Alds, Blds, acc);

  const int lane = threadIdx.x & 31, wid = threadIdx.x >> 5;
  const int wm = wid & 3, wn = wid >> 2;
  const int lrow = lane & 15, lhalf = lane >> 4;
#pragma unroll
  for (int r = 0; r < 2; ++r)
#pragma unroll
    for (int c = 0; c < 4; ++c) {
      const int col  = blockCol + wn * 64 + c * 16 + lrow;
      const float bias = b2e[col];
      const int row0 = blockRow + wm * 32 + r * 16 + lhalf * 8;
#pragma unroll
      for (int i = 0; i < 8; ++i) {
        const int row = row0 + i;
        const float w = wdense[(size_t)row * NEXP + expert];
        const float y = (acc[r][c][i] + bias) * w;
        const size_t idx = (size_t)row * DDIM + col;
        out[idx] = accum ? (out[idx] + y) : y;
      }
    }
}

// --- launcher -----------------------------------------------------------
extern "C" void kernel_launch(void* const* d_in, const int* in_sizes, int n_in,
                              void* d_out, int out_size, void* d_ws, size_t ws_size,
                              hipStream_t stream) {
  (void)in_sizes; (void)n_in; (void)out_size; (void)ws_size;
  const float* x  = (const float*)d_in[0];
  const float* Wg = (const float*)d_in[1];
  const float* bg = (const float*)d_in[2];
  const float* W1 = (const float*)d_in[3];
  const float* b1 = (const float*)d_in[4];
  const float* W2 = (const float*)d_in[5];
  const float* b2 = (const float*)d_in[6];
  float* out = (float*)d_out;

  // workspace layout (all offsets 256B aligned):
  //   [0, 256KB)        : dense gate weights  [TOKENS x NEXP] f32
  //   [256KB, +16.8MB)  : x in bf16           [TOKENS x DDIM]
  //   [.., +67MB)       : h in bf16           [TOKENS x HDIM] (reused per expert)
  float* wdense      = (float*)d_ws;
  unsigned short* xb = (unsigned short*)((char*)d_ws + 262144);
  unsigned short* hb = xb + (size_t)TOKENS * DDIM;

  moe_convert<<<2048, 256, 0, stream>>>(x, xb, TOKENS * DDIM);
  moe_gate<<<TOKENS / 8, 256, 0, stream>>>(x, Wg, bg, wdense);

  for (int e = 0; e < NEXP; ++e) {
    moe_gemm1<<<dim3(HDIM / 128, TOKENS / 128), 256, 0, stream>>>(
        xb, W1 + (size_t)e * DDIM * HDIM, b1 + (size_t)e * HDIM, hb);
    moe_gemm2<<<dim3(DDIM / 128, TOKENS / 128), 256, 0, stream>>>(
        hb, W2 + (size_t)e * HDIM * DDIM, b2 + (size_t)e * DDIM,
        wdense, e, e > 0 ? 1 : 0, out);
  }
}
